// SingleDiffPool_55439437857008
// MI455X (gfx1250) — matrix operand order repaired and checked by reading, not compiled, not run on hardware
//
#include <hip/hip_runtime.h>
#include <hip/hip_bf16.h>
#include <math.h>

#define B_GR 8
#define N_NODES 2000
#define FEA 64
#define E_PER 64000
#define C1 200
#define NEG_SLOPE 0.2f

typedef __attribute__((ext_vector_type(16))) __bf16 v16bf;
typedef __attribute__((ext_vector_type(8)))  float  v8f;

// ---------------- small utility kernels ----------------
__global__ void k_fill_f32(float* p, float v, long n) {
  long i = (long)blockIdx.x * blockDim.x + threadIdx.x;
  if (i < n) p[i] = v;
}
__global__ void k_fill_u32(unsigned* p, unsigned v, long n) {
  long i = (long)blockIdx.x * blockDim.x + threadIdx.x;
  if (i < n) p[i] = v;
}

__device__ inline unsigned enc_f32(float f) {
  unsigned u = __float_as_uint(f);
  return (u & 0x80000000u) ? ~u : (u | 0x80000000u);
}
__device__ inline float dec_f32(unsigned u) {
  u = (u & 0x80000000u) ? (u & 0x7fffffffu) : ~u;
  return __uint_as_float(u);
}

__device__ inline v8f wmma_bf16(const v16bf& a, const v16bf& b, const v8f& c) {
  return __builtin_amdgcn_wmma_f32_16x16x32_bf16(false, a, false, b, (short)0, c,
                                                 false, false);
}

// ---------------- feature GEMM: Y = X(M x 64) @ W(64 x fout) + bias ----------------
// M must be a multiple of 16. 256 threads = 8 waves; X tile + full W staged in LDS,
// branch-free float4 staging, K=64 fully unrolled (2 WMMAs per wave per n-tile).
// Out-of-range output columns are handled by CLAMPING the LDS column index (a WMMA
// B-column only feeds the same D-column, so garbage columns are never stored).
__global__ void k_feat_gemm(const float* __restrict__ X, const float* __restrict__ W,
                            const float* __restrict__ bias, float* __restrict__ Y,
                            int M, int fout) {
  __shared__ float lX[16][68];      // 16 x 64 (pad 4)
  __shared__ float lW[64 * 201];    // 64 x fout (pad 1), max fout = 200
  const int tid = threadIdx.x;
  const int wave = tid >> 5, lane = tid & 31;
  const int m0 = blockIdx.x * 16;
  const int ldw = fout + 1;
  {  // stage X tile: one float4 per thread, fully coalesced, no branches
    int row = tid >> 4, c4 = (tid & 15) * 4;
    const float4 v = *(const float4*)(X + (long)(m0 + row) * FEA + c4);
    lX[row][c4] = v.x; lX[row][c4 + 1] = v.y; lX[row][c4 + 2] = v.z; lX[row][c4 + 3] = v.w;
  }
  {  // stage all of W, division-free: 4 threads per row
    const int k = tid >> 2;                       // 0..63
    const float* wrow = W + k * fout;
    float* lrow = lW + k * ldw;
    for (int n = (tid & 3); n < fout; n += 4) lrow[n] = wrow[n];
  }
  __syncthreads();
  const int half = lane >> 4, l15 = lane & 15;
  // A fragments for both K-steps (CDNA5 16-bit A layout)
  v16bf a0{}, a1{};
#pragma unroll
  for (int t = 0; t < 8; ++t) {
    a0[t]     = (__bf16)lX[l15][half * 8 + t];
    a0[8 + t] = (__bf16)lX[l15][16 + half * 8 + t];
    a1[t]     = (__bf16)lX[l15][32 + half * 8 + t];
    a1[8 + t] = (__bf16)lX[l15][48 + half * 8 + t];
  }
  const int ntiles = (fout + 15) / 16;
  for (int nt = wave; nt < ntiles; nt += 8) {
    const int gn = nt * 16 + l15;
    const bool on = gn < fout;
    const int cn = on ? gn : (fout - 1);          // clamped column, branch-free loads
    v16bf b0{}, b1{};
#pragma unroll
    for (int t = 0; t < 16; ++t) {
      b0[t] = (__bf16)lW[((t < 8 ? half * 8 + t : 16 + half * 8 + (t - 8))) * ldw + cn];
      b1[t] = (__bf16)lW[(32 + (t < 8 ? half * 8 + t : 16 + half * 8 + (t - 8))) * ldw + cn];
    }
    v8f acc = {};
    acc = wmma_bf16(a0, b0, acc);
    acc = wmma_bf16(a1, b1, acc);
    if (on) {
      float bv = bias[gn];
#pragma unroll
      for (int j = 0; j < 8; ++j)
        Y[(long)(m0 + half * 8 + j) * fout + gn] = acc[j] + bv;
    }
  }
}

// ---------------- generic WMMA GEMM (used for pooled S^T @ Z) ----------------
template <bool TA, bool TB>
__global__ void k_wmma_gemm(const float* __restrict__ A, const float* __restrict__ Bm,
                            float* __restrict__ C, const float* __restrict__ bias,
                            int M, int N, int K, int lda, int ldb, int ldc,
                            long sA, long sB, long sC) {
  __shared__ float lA[16][33];
  __shared__ float lB[32][17];
  const int lane = threadIdx.x;
  const int m0 = blockIdx.x * 16, n0 = blockIdx.y * 16;
  A  += (long)blockIdx.z * sA;
  Bm += (long)blockIdx.z * sB;
  C  += (long)blockIdx.z * sC;
  const int half = lane >> 4, l15 = lane & 15;
  v8f acc = {};
  for (int k0 = 0; k0 < K; k0 += 32) {
    for (int i = lane; i < 16 * 32; i += 32) {
      int m = i >> 5, k = i & 31;
      int gm = min(m0 + m, M - 1), gk = k0 + k;   // clamp rows, zero-fill K
      float v = 0.f;
      if (gk < K) v = TA ? A[(long)gk * lda + gm] : A[(long)gm * lda + gk];
      lA[m][k] = v;
    }
    for (int i = lane; i < 32 * 16; i += 32) {
      int k = i >> 4, n = i & 15;
      int gk = k0 + k, gn = min(n0 + n, N - 1);   // clamp cols, zero-fill K
      float v = 0.f;
      if (gk < K) v = TB ? Bm[(long)gn * ldb + gk] : Bm[(long)gk * ldb + gn];
      lB[k][n] = v;
    }
    __syncthreads();
    v16bf af{}, bfr{};
#pragma unroll
    for (int t = 0; t < 8; ++t) {
      af[t]     = (__bf16)lA[l15][half * 8 + t];
      af[8 + t] = (__bf16)lA[l15][16 + half * 8 + t];
    }
#pragma unroll
    for (int t = 0; t < 16; ++t) bfr[t] = (__bf16)lB[t < 8 ? half * 8 + t : 16 + half * 8 + (t - 8)][l15];
    acc = wmma_bf16(af, bfr, acc);
    __syncthreads();
  }
  int gn = n0 + l15;
  if (gn < N) {
    float bv = bias ? bias[gn] : 0.f;
#pragma unroll
    for (int j = 0; j < 8; ++j) {
      int gm = m0 + half * 8 + j;
      if (gm < M) C[(long)gm * ldc + gn] = acc[j] + bv;
    }
  }
}

// ---------------- fused link loss: sum (adj - S S^T)^2 ----------------
// 256 threads = 8 waves as 2x4 tile grid -> 32(n) x 64(m) region per block.
__global__ void k_link_loss(const float* __restrict__ S, const float* __restrict__ adj,
                            int nn, int C, float* __restrict__ accum) {
  __shared__ float lA[32][36];   // 32 n-rows x 32 k
  __shared__ float lB[32][68];   // 32 k x 64 m-cols (S^T)
  const int tid = threadIdx.x, wave = tid >> 5, lane = tid & 31;
  const int n0 = blockIdx.x * 32, m0 = blockIdx.y * 64;
  const float* Sb   = S   + (long)blockIdx.z * nn * C;
  const float* adjb = adj + (long)blockIdx.z * nn * nn;
  const int wr = wave >> 2, wc = wave & 3;
  const int tn0 = n0 + wr * 16, tm0 = m0 + wc * 16;
  const int half = lane >> 4, l15 = lane & 15;
  {  // prefetch the adj tile we will need after the K loop (gfx1250 global_prefetch)
    int r = tid & 31, part = tid >> 5;
    int gr = n0 + r, gc = m0 + part * 8;
    if (gr < nn && gc < nn) __builtin_prefetch(&adjb[(long)gr * nn + gc], 0, 3);
  }
  v8f acc = {};
  for (int k0 = 0; k0 < C; k0 += 32) {
    for (int i = tid; i < 32 * 32; i += 256) {       // A: S rows n0..n0+31
      int r = i >> 5, k = i & 31;
      int gr = min(n0 + r, nn - 1), gk = k0 + k;     // clamp rows, zero-fill K
      lA[r][k] = (gk < C) ? Sb[(long)gr * C + gk] : 0.f;
    }
    for (int i = tid; i < 64 * 32; i += 256) {       // B: S rows m0..m0+63, transposed
      int c = i >> 5, k = i & 31;
      int gm = min(m0 + c, nn - 1), gk = k0 + k;
      lB[k][c] = (gk < C) ? Sb[(long)gm * C + gk] : 0.f;
    }
    __syncthreads();
    v16bf af{}, bfr{};
    const int ar = wr * 16 + l15, bc = wc * 16 + l15;
#pragma unroll
    for (int t = 0; t < 8; ++t) {
      af[t]     = (__bf16)lA[ar][half * 8 + t];
      af[8 + t] = (__bf16)lA[ar][16 + half * 8 + t];
    }
#pragma unroll
    for (int t = 0; t < 16; ++t) bfr[t] = (__bf16)lB[t < 8 ? half * 8 + t : 16 + half * 8 + (t - 8)][bc];
    acc = wmma_bf16(af, bfr, acc);
    __syncthreads();
  }
  float local = 0.f;
  const int gm = tm0 + l15;
  if (gm < nn) {
#pragma unroll
    for (int j = 0; j < 8; ++j) {
      int gn = tn0 + half * 8 + j;
      if (gn < nn) {
        float d = adjb[(long)gn * nn + gm] - acc[j];
        local += d * d;
      }
    }
  }
  for (int off = 16; off > 0; off >>= 1) local += __shfl_xor(local, off, 32);
  if (lane == 0) atomicAdd(accum, local);
}

// ---------------- GATv2 edge kernels ----------------
__global__ void k_edge_scores(const float* __restrict__ xl, const float* __restrict__ xr,
                              const float* __restrict__ att, const int* __restrict__ src,
                              const int* __restrict__ dst, int E, int fout,
                              float* __restrict__ e) {
  int i = blockIdx.x * blockDim.x + threadIdx.x;
  if (i >= E) return;
  const float* pl = xl + (long)src[i] * fout;
  const float* pr = xr + (long)dst[i] * fout;
  float acc = 0.f;
  for (int k = 0; k < fout; ++k) {
    float v = pl[k] + pr[k];
    v = v > 0.f ? v : NEG_SLOPE * v;
    acc += att[k] * v;
  }
  e[i] = acc;
}
__global__ void k_edge_max(const float* e, const int* dst, unsigned* menc, int E) {
  int i = blockIdx.x * blockDim.x + threadIdx.x;
  if (i < E) atomicMax(&menc[dst[i]], enc_f32(e[i]));
}
__global__ void k_decode_max(const unsigned* menc, float* m, int n) {
  int i = blockIdx.x * blockDim.x + threadIdx.x;
  if (i < n) m[i] = dec_f32(menc[i]);
}
__global__ void k_edge_expsum(const float* e, const float* m, const int* dst, float* w,
                              float* denom, int E) {
  int i = blockIdx.x * blockDim.x + threadIdx.x;
  if (i >= E) return;
  float wv = __expf(e[i] - m[dst[i]]);
  w[i] = wv;
  atomicAdd(&denom[dst[i]], wv);
}
__global__ void k_init_bias(float* out, const float* bias, long n, int fout) {
  long i = (long)blockIdx.x * blockDim.x + threadIdx.x;
  if (i < n * fout) out[i] = bias[i % fout];
}
__global__ void k_edge_agg(const float* w, const float* denom, const float* xl,
                           const int* src, const int* dst, float* out, int E, int fout) {
  int i = blockIdx.x * blockDim.x + threadIdx.x;
  if (i >= E) return;
  int d = dst[i];
  float a = w[i] / (denom[d] + 1e-16f);
  const float* pl = xl + (long)src[i] * fout;
  float* po = out + (long)d * fout;
  for (int k = 0; k < fout; ++k) atomicAdd(&po[k], a * pl[k]);
}

// ---------------- dense helpers ----------------
__global__ void k_softmax_rows(float* s, int n, int C) {
  int i = blockIdx.x * blockDim.x + threadIdx.x;
  if (i >= n) return;
  float* row = s + (long)i * C;
  float mx = -INFINITY;
  for (int c = 0; c < C; ++c) mx = fmaxf(mx, row[c]);
  float sum = 0.f;
  for (int c = 0; c < C; ++c) { float v = __expf(row[c] - mx); row[c] = v; sum += v; }
  float inv = 1.f / sum;
  for (int c = 0; c < C; ++c) row[c] *= inv;
}
__global__ void k_adj_scatter(const int* src, const int* dst, float* adj, int cur, int E) {
  int i = blockIdx.x * blockDim.x + threadIdx.x;
  if (i >= E) return;
  int s = src[i], d = dst[i], b = s / cur;
  atomicAdd(&adj[((long)b * cur + (s - b * cur)) * cur + (d % cur)], 1.0f);
}
__global__ void k_entropy(const float* s, int n, int C, float* accum) {
  int i = blockIdx.x * blockDim.x + threadIdx.x;
  if (i >= n) return;
  const float* row = s + (long)i * C;
  float acc = 0.f;
  for (int c = 0; c < C; ++c) { float p = row[c]; acc += -p * __logf(p + 1e-15f); }
  atomicAdd(accum, acc);
}
__global__ void k_gen_edges(int* src, int* dst, int C, int E) {
  int i = blockIdx.x * blockDim.x + threadIdx.x;
  if (i >= E) return;
  int per = C * C, b = i / per, r = i - b * per;
  src[i] = b * C + r / C;
  dst[i] = b * C + r % C;
}
__global__ void k_cluster_head(const float* xd, const float* W, const float* bb, int C,
                               float* dest) {
  int i = blockIdx.x * blockDim.x + threadIdx.x;
  if (i >= B_GR * 2) return;
  int b = i >> 1, j = i & 1;
  float acc = bb[j];
  for (int k = 0; k < FEA; ++k) {
    float mean = 0.f;
    for (int c = 0; c < C; ++c) mean += xd[((long)b * C + c) * FEA + k];
    acc += (mean / (float)C) * W[k * 2 + j];
  }
  dest[i] = acc;
}
__global__ void k_final_linear(const float* x, const float* W, const float* bb, int rows,
                               float* dest) {
  int i = blockIdx.x * blockDim.x + threadIdx.x;
  if (i >= rows * 2) return;
  int r = i >> 1, j = i & 1;
  float acc = bb[j];
  for (int k = 0; k < FEA; ++k) acc += x[(long)r * FEA + k] * W[k * 2 + j];
  dest[i] = acc;
}
__global__ void k_finalize(const float* accum, float* dout) {
  if (threadIdx.x == 0) {
    dout[16] = sqrtf(accum[0]) / ((float)B_GR * N_NODES * N_NODES);
    dout[17] = sqrtf(accum[1]) / ((float)B_GR * C1 * C1);
    dout[18] = accum[2] / ((float)B_GR * N_NODES);
    dout[19] = accum[3] / ((float)B_GR * C1);
  }
}

// ---------------- host orchestration ----------------
struct GatP { const float *Wl, *bl, *Wr, *br, *att, *bias; int fout; };

static void run_gat(const float* x, int n, const GatP& p, const int* src, const int* dst,
                    int E, float* xl, float* xr, float* e, float* w, unsigned* menc,
                    float* m, float* denom, float* out, hipStream_t st) {
  const int fo = p.fout;
  k_feat_gemm<<<n / 16, 256, 0, st>>>(x, p.Wl, p.bl, xl, n, fo);
  k_feat_gemm<<<n / 16, 256, 0, st>>>(x, p.Wr, p.br, xr, n, fo);
  int eb = (E + 255) / 256, nb = (n + 255) / 256;
  k_edge_scores<<<eb, 256, 0, st>>>(xl, xr, p.att, src, dst, E, fo, e);
  k_fill_u32<<<nb, 256, 0, st>>>(menc, 0x007FFFFFu, n);  // enc(-inf)
  k_edge_max<<<eb, 256, 0, st>>>(e, dst, menc, E);
  k_decode_max<<<nb, 256, 0, st>>>(menc, m, n);
  k_fill_f32<<<nb, 256, 0, st>>>(denom, 0.f, n);
  k_edge_expsum<<<eb, 256, 0, st>>>(e, m, dst, w, denom, E);
  long tot = (long)n * fo;
  k_init_bias<<<(unsigned)((tot + 255) / 256), 256, 0, st>>>(out, p.bias, n, fo);
  k_edge_agg<<<eb, 256, 0, st>>>(w, denom, xl, src, dst, out, E, fo);
}

extern "C" void kernel_launch(void* const* d_in, const int* in_sizes, int n_in,
                              void* d_out, int out_size, void* d_ws, size_t ws_size,
                              hipStream_t stream) {
  const float* x0 = (const float*)d_in[0];
  const int* eidx = (const int*)d_in[1];
  const int E1 = B_GR * E_PER;
  const int* src1 = eidx;
  const int* dst1 = eidx + E1;
  auto gat = [&](int b) {
    GatP g;
    g.Wl = (const float*)d_in[b];     g.bl = (const float*)d_in[b + 1];
    g.Wr = (const float*)d_in[b + 2]; g.br = (const float*)d_in[b + 3];
    g.att = (const float*)d_in[b + 4]; g.bias = (const float*)d_in[b + 5];
    g.fout = in_sizes[b + 1];
    return g;
  };
  GatP p00 = gat(3),  p01 = gat(9),  p10 = gat(15), p11 = gat(21);
  GatP q00 = gat(27), q01 = gat(33), q10 = gat(39), q11 = gat(45);
  const float* headW0 = (const float*)d_in[51];
  const float* headW1 = (const float*)d_in[52];
  const float* headB0 = (const float*)d_in[53];
  const float* headB1 = (const float*)d_in[54];
  const float* linW = (const float*)d_in[55];
  const float* linB = (const float*)d_in[56];
  float* dout = (float*)d_out;

  char* wp = (char*)d_ws;
  auto carve = [&](size_t bytes) {
    void* r = (void*)wp;
    wp += (bytes + 255) & ~(size_t)255;
    return r;
  };
  const int NB1 = B_GR * N_NODES;                 // 16000
  const int NB2 = B_GR * C1;                      // 1600
  const int E2 = B_GR * C1 * C1;                  // 320000
  float*    xl    = (float*)carve((size_t)NB1 * C1 * 4);
  float*    xr    = (float*)carve((size_t)NB1 * C1 * 4);
  float*    ebuf  = (float*)carve((size_t)E1 * 4);
  float*    wbuf  = (float*)carve((size_t)E1 * 4);
  unsigned* menc  = (unsigned*)carve((size_t)NB1 * 4);
  float*    mbuf  = (float*)carve((size_t)NB1 * 4);
  float*    denom = (float*)carve((size_t)NB1 * 4);
  float*    t0    = (float*)carve((size_t)NB1 * FEA * 4);
  float*    sbuf  = (float*)carve((size_t)NB1 * C1 * 4);
  float*    z0    = (float*)carve((size_t)NB1 * FEA * 4);
  float*    z1    = (float*)carve((size_t)NB1 * FEA * 4);
  float*    adj   = (float*)carve((size_t)B_GR * N_NODES * N_NODES * 4);
  float*    xd    = (float*)carve((size_t)B_GR * C1 * FEA * 4);
  float*    xd2   = (float*)carve((size_t)B_GR * FEA * 4);
  int*      src2  = (int*)carve((size_t)E2 * 4);
  int*      dst2  = (int*)carve((size_t)E2 * 4);
  float*    accum = (float*)carve(4 * 4);

  k_fill_f32<<<1, 32, 0, stream>>>(accum, 0.f, 4);

  // ===== pooling layer 0 (cur = 2000) =====
  run_gat(x0, NB1, p00, src1, dst1, E1, xl, xr, ebuf, wbuf, menc, mbuf, denom, t0, stream);
  run_gat(t0, NB1, p01, src1, dst1, E1, xl, xr, ebuf, wbuf, menc, mbuf, denom, sbuf, stream);
  run_gat(x0, NB1, q00, src1, dst1, E1, xl, xr, ebuf, wbuf, menc, mbuf, denom, z0, stream);
  run_gat(z0, NB1, q01, src1, dst1, E1, xl, xr, ebuf, wbuf, menc, mbuf, denom, z1, stream);
  long adjn1 = (long)B_GR * N_NODES * N_NODES;
  k_fill_f32<<<(unsigned)((adjn1 + 255) / 256), 256, 0, stream>>>(adj, 0.f, adjn1);
  k_adj_scatter<<<(E1 + 255) / 256, 256, 0, stream>>>(src1, dst1, adj, N_NODES, E1);
  k_softmax_rows<<<(NB1 + 255) / 256, 256, 0, stream>>>(sbuf, NB1, C1);
  k_entropy<<<(NB1 + 255) / 256, 256, 0, stream>>>(sbuf, NB1, C1, accum + 2);
  dim3 lg((N_NODES + 31) / 32, (N_NODES + 63) / 64, B_GR);
  k_link_loss<<<lg, 256, 0, stream>>>(sbuf, adj, N_NODES, C1, accum + 0);
  dim3 pg((C1 + 15) / 16, FEA / 16, B_GR);   // xd[b] = S_b^T @ Z_b
  k_wmma_gemm<true, false><<<pg, 32, 0, stream>>>(sbuf, z1, xd, nullptr, C1, FEA, N_NODES,
      C1, FEA, FEA, (long)N_NODES * C1, (long)N_NODES * FEA, (long)C1 * FEA);
  k_cluster_head<<<1, 32, 0, stream>>>(xd, headW0, headB0, C1, dout + 20);

  // ===== pooling layer 1 (cur = 200, Cl = 1, complete graph) =====
  k_gen_edges<<<(E2 + 255) / 256, 256, 0, stream>>>(src2, dst2, C1, E2);
  run_gat(xd, NB2, p10, src2, dst2, E2, xl, xr, ebuf, wbuf, menc, mbuf, denom, t0, stream);
  run_gat(t0, NB2, p11, src2, dst2, E2, xl, xr, ebuf, wbuf, menc, mbuf, denom, sbuf, stream);
  run_gat(xd, NB2, q10, src2, dst2, E2, xl, xr, ebuf, wbuf, menc, mbuf, denom, z0, stream);
  run_gat(z0, NB2, q11, src2, dst2, E2, xl, xr, ebuf, wbuf, menc, mbuf, denom, z1, stream);
  long adjn2 = (long)B_GR * C1 * C1;
  k_fill_f32<<<(unsigned)((adjn2 + 255) / 256), 256, 0, stream>>>(adj, 0.f, adjn2);
  k_adj_scatter<<<(E2 + 255) / 256, 256, 0, stream>>>(src2, dst2, adj, C1, E2);
  k_softmax_rows<<<(NB2 + 255) / 256, 256, 0, stream>>>(sbuf, NB2, 1);
  k_entropy<<<(NB2 + 255) / 256, 256, 0, stream>>>(sbuf, NB2, 1, accum + 3);
  dim3 lg2((C1 + 31) / 32, (C1 + 63) / 64, B_GR);
  k_link_loss<<<lg2, 256, 0, stream>>>(sbuf, adj, C1, 1, accum + 1);
  dim3 pg2(1, FEA / 16, B_GR);               // xd2[b] = s2_b^T @ Z_b  (1 x 64)
  k_wmma_gemm<true, false><<<pg2, 32, 0, stream>>>(sbuf, z1, xd2, nullptr, 1, FEA, C1,
      1, FEA, FEA, (long)C1, (long)C1 * FEA, (long)FEA);
  k_cluster_head<<<1, 32, 0, stream>>>(xd2, headW1, headB1, 1, dout + 36);

  // ===== heads / outputs =====
  k_final_linear<<<1, 32, 0, stream>>>(xd2, linW, linB, B_GR, dout);
  k_finalize<<<1, 32, 0, stream>>>(accum, dout);
}
